// EBM_GL_65884798320752
// MI455X (gfx1250) — matrix operand order repaired and checked by reading, not compiled, not run on hardware
//
#include <hip/hip_runtime.h>
#include <hip/hip_bf16.h>
#include <stdint.h>

// ---------------------------------------------------------------------------
// Types for CDNA5 WMMA
// ---------------------------------------------------------------------------
typedef __attribute__((ext_vector_type(16))) __bf16 v16bf;
typedef __attribute__((ext_vector_type(8)))  float  v8f;

#define ROTL32(x,n) (((x) << (n)) | ((x) >> (32 - (n))))

// Threefry-2x32-20, exactly as used by JAX's PRNG.
__host__ __device__ inline void threefry2x32(uint32_t k0, uint32_t k1,
                                             uint32_t c0, uint32_t c1,
                                             uint32_t* o0, uint32_t* o1) {
  uint32_t ks2 = 0x1BD11BDAu ^ k0 ^ k1;
  uint32_t x0 = c0 + k0, x1 = c1 + k1;
#define TF_R(r) { x0 += x1; x1 = ROTL32(x1, r); x1 ^= x0; }
  TF_R(13) TF_R(15) TF_R(26) TF_R(6)  x0 += k1;  x1 += ks2 + 1u;
  TF_R(17) TF_R(29) TF_R(16) TF_R(24) x0 += ks2; x1 += k0  + 2u;
  TF_R(13) TF_R(15) TF_R(26) TF_R(6)  x0 += k0;  x1 += k1  + 3u;
  TF_R(17) TF_R(29) TF_R(16) TF_R(24) x0 += k1;  x1 += ks2 + 4u;
  TF_R(13) TF_R(15) TF_R(26) TF_R(6)  x0 += ks2; x1 += k0  + 5u;
#undef TF_R
  *o0 = x0; *o1 = x1;
}

// jax.random.bernoulli(key, 0.8, (256,1024)) dropout: element e of 262144.
__device__ inline float jax_dropout(float v, uint32_t key0, uint32_t key1,
                                    uint32_t e) {
  const uint32_t HALF = 131072u;  // 256*1024/2
  uint32_t o0, o1, bits;
  if (e < HALF) { threefry2x32(key0, key1, e, e + HALF, &o0, &o1); bits = o0; }
  else          { threefry2x32(key0, key1, e - HALF, e, &o0, &o1); bits = o1; }
  float u = __uint_as_float((bits >> 9) | 0x3f800000u) - 1.0f;
  return (u < 0.8f) ? v * 1.25f : 0.0f;  // keep-prob 0.8, scale 1/0.8
}

// ---------------------------------------------------------------------------
// K0: inverse L2 norms of y_embedding rows (200) and c_embedding rows (224)
// ---------------------------------------------------------------------------
__global__ __launch_bounds__(256) void norms_kernel(
    const float* __restrict__ y_emb, const float* __restrict__ c_emb,
    float* __restrict__ inv_ny, float* __restrict__ inv_nc) {
  int wid  = (blockIdx.x * blockDim.x + threadIdx.x) >> 5;  // 0..423
  int lane = threadIdx.x & 31;
  if (wid >= 424) return;
  const float* row; float* out; int idx;
  if (wid < 200) { row = y_emb + (size_t)wid * 1024; out = inv_ny; idx = wid; }
  else { idx = wid - 200; row = c_emb + (size_t)idx * 1024; out = inv_nc; }
  float s = 0.f;
  for (int h = lane * 4; h < 1024; h += 128) {
    float4 v = *(const float4*)(row + h);
    s += v.x * v.x + v.y * v.y + v.z * v.z + v.w * v.w;
  }
  for (int off = 16; off; off >>= 1) s += __shfl_xor(s, off, 32);
  if (lane == 0) out[idx] = 1.0f / fmaxf(sqrtf(s), 1e-12f);
}

// ---------------------------------------------------------------------------
// K1: h = dropout(x @ W.T + b) via v_wmma_f32_16x16x32_bf16.
// One wave per 16x16 output tile; K unrolled x2 with independent accumulator
// chains so two WMMAs can be in flight per wave (C-operand chain is serial
// otherwise). Fragment loads are explicit b128.
// grid = (16, 64, 2); z selects xy/xc head.
// ---------------------------------------------------------------------------
__device__ inline void load_frags(const float* __restrict__ xrow,
                                  const float* __restrict__ wrow,
                                  int k0, int half, v16bf* amat, v16bf* bmat) {
#pragma unroll
  for (int g = 0; g < 2; ++g) {
    // A 16x32 bf16 layout: element e=8g+t  ->  K = 16g + 8*half + t, M = lm
    const float4 a0 = *(const float4*)(xrow + k0 + 16 * g + 8 * half);
    const float4 a1 = *(const float4*)(xrow + k0 + 16 * g + 8 * half + 4);
    (*amat)[8 * g + 0] = (__bf16)a0.x; (*amat)[8 * g + 1] = (__bf16)a0.y;
    (*amat)[8 * g + 2] = (__bf16)a0.z; (*amat)[8 * g + 3] = (__bf16)a0.w;
    (*amat)[8 * g + 4] = (__bf16)a1.x; (*amat)[8 * g + 5] = (__bf16)a1.y;
    (*amat)[8 * g + 6] = (__bf16)a1.z; (*amat)[8 * g + 7] = (__bf16)a1.w;
  }
#pragma unroll
  for (int q = 0; q < 4; ++q) {
    // B 32x16 bf16 layout: element e=8g+t -> K = 16g + 8*half + t, N = lm (B[k][n]=W[n][k])
    const float4 b0 = *(const float4*)(wrow + k0 + 16 * (q >> 1) + 8 * half + 4 * (q & 1));
    (*bmat)[4 * q + 0] = (__bf16)b0.x; (*bmat)[4 * q + 1] = (__bf16)b0.y;
    (*bmat)[4 * q + 2] = (__bf16)b0.z; (*bmat)[4 * q + 3] = (__bf16)b0.w;
  }
}

__global__ __launch_bounds__(32) void fc1_wmma_kernel(
    const float* __restrict__ x,
    const float* __restrict__ w_xy, const float* __restrict__ b_xy,
    const float* __restrict__ w_xc, const float* __restrict__ b_xc,
    float* __restrict__ h_xy, float* __restrict__ h_xc,
    uint32_t k1a, uint32_t k1b, uint32_t k2a, uint32_t k2b) {
  const int which = blockIdx.z;
  const float* __restrict__ W    = which ? w_xc : w_xy;
  const float* __restrict__ bias = which ? b_xc : b_xy;
  float* __restrict__ out        = which ? h_xc : h_xy;
  const uint32_t key0 = which ? k2a : k1a;
  const uint32_t key1 = which ? k2b : k1b;

  const int lane = threadIdx.x;
  const int half = lane >> 4;      // 0: lanes 0-15, 1: lanes 16-31
  const int lm   = lane & 15;
  const int m0 = blockIdx.x * 16;
  const int n0 = blockIdx.y * 16;

  const float* __restrict__ xrow = x + (size_t)(m0 + lm) * 2048;
  const float* __restrict__ wrow = W + (size_t)(n0 + lm) * 2048;

  v8f acc0 = {}, acc1 = {};
  for (int k0 = 0; k0 < 2048; k0 += 64) {
    v16bf a0, b0, a1, b1;
    load_frags(xrow, wrow, k0,      half, &a0, &b0);
    load_frags(xrow, wrow, k0 + 32, half, &a1, &b1);
    acc0 = __builtin_amdgcn_wmma_f32_16x16x32_bf16(false, a0, false, b0,
                                                   (short)0, acc0, false, false);
    acc1 = __builtin_amdgcn_wmma_f32_16x16x32_bf16(false, a1, false, b1,
                                                   (short)0, acc1, false, false);
  }
  const int n = n0 + lm;
  const float bn = bias[n];
#pragma unroll
  for (int r = 0; r < 8; ++r) {
    const int m = m0 + r + (half << 3);  // C/D: row = r + 8*half, col = lm
    float v = acc0[r] + acc1[r] + bn;
    out[(size_t)m * 1024 + n] = jax_dropout(v, key0, key1, (uint32_t)(m * 1024 + n));
  }
}

// ---------------------------------------------------------------------------
// K2: per-concept GEMV pair against the big projection weight (the 470MB
// HBM stream -> this kernel IS the roofline cost, ~20us at 23.3 TB/s).
// P[n,h'] = c_emb[n] . W[h', n*H:...], Q likewise for the negative row.
// Stores D = P - Q and Q. One wave per (n, h'); 4KB contiguous W slice per
// wave, read once, dotted against both embeddings.
// ---------------------------------------------------------------------------
__global__ __launch_bounds__(256) void pq_kernel(
    const float* __restrict__ c_emb, const float* __restrict__ W,
    float* __restrict__ Dm, float* __restrict__ Qm) {
  const int n    = blockIdx.x;             // 0..111
  const int wave = threadIdx.x >> 5;
  const int lane = threadIdx.x & 31;
  const int hp   = blockIdx.y * 8 + wave;  // 0..1023
  const float* __restrict__ wrow = W + (size_t)hp * 114688 + (size_t)n * 1024;
  const float* __restrict__ cp = c_emb + (size_t)n * 1024;
  const float* __restrict__ cn = c_emb + (size_t)(112 + n) * 1024;
  float sp = 0.f, sn = 0.f;
  for (int h = lane * 4; h < 1024; h += 128) {
    __builtin_prefetch(wrow + h + 128);   // global_prefetch_b8
    float4 wv = *(const float4*)(wrow + h);
    float4 pv = *(const float4*)(cp + h);
    float4 nv = *(const float4*)(cn + h);
    sp += wv.x * pv.x + wv.y * pv.y + wv.z * pv.z + wv.w * pv.w;
    sn += wv.x * nv.x + wv.y * nv.y + wv.z * nv.z + wv.w * nv.w;
  }
  for (int off = 16; off; off >>= 1) {
    sp += __shfl_xor(sp, off, 32);
    sn += __shfl_xor(sn, off, 32);
  }
  if (lane == 0) {
    Dm[(size_t)n * 1024 + hp] = sp - sn;
    Qm[(size_t)n * 1024 + hp] = sn;
  }
}

// ---------------------------------------------------------------------------
// K3: Qsum[h] = concept_proj_b[h] + sum_n Q[n,h]
// ---------------------------------------------------------------------------
__global__ __launch_bounds__(256) void qsum_kernel(
    const float* __restrict__ Qm, const float* __restrict__ bias,
    float* __restrict__ qsum) {
  int h = blockIdx.x * 256 + threadIdx.x;
  if (h < 1024) {
    float s = bias[h];
    for (int n = 0; n < 112; ++n) s += Qm[(size_t)n * 1024 + h];
    qsum[h] = s;
  }
}

// ---------------------------------------------------------------------------
// K4: c_proj[b,h] = Qsum[h] + sum_{n: c_gt[b,n]==1} D[n,h].  One block per b;
// c_gt[b,n] is uniform across the block -> scalar branch, L2-resident D.
// ---------------------------------------------------------------------------
__global__ __launch_bounds__(256) void cproj_kernel(
    const int* __restrict__ c_gt, const float* __restrict__ Dm,
    const float* __restrict__ qsum, float* __restrict__ c_proj) {
  const int b = blockIdx.x;
  const int t = threadIdx.x;
  float acc0 = qsum[t], acc1 = qsum[t + 256], acc2 = qsum[t + 512], acc3 = qsum[t + 768];
  const int* __restrict__ g = c_gt + (size_t)b * 112;
  for (int n = 0; n < 112; ++n) {
    if (g[n] == 1) {
      const float* __restrict__ d = Dm + (size_t)n * 1024;
      acc0 += d[t]; acc1 += d[t + 256]; acc2 += d[t + 512]; acc3 += d[t + 768];
    }
  }
  float* __restrict__ o = c_proj + (size_t)b * 1024;
  o[t] = acc0; o[t + 256] = acc1; o[t + 512] = acc2; o[t + 768] = acc3;
}

// ---------------------------------------------------------------------------
// K5: fused energy reductions. One wave per output scalar.
// j in [0,200): xy_energy ; [200,400): cy_energy ; [400,624): xc_energy(n,s)
// ---------------------------------------------------------------------------
__global__ __launch_bounds__(256) void energy_kernel(
    const float* __restrict__ h_xy, const float* __restrict__ h_xc,
    const float* __restrict__ c_proj,
    const float* __restrict__ y_emb, const float* __restrict__ c_emb,
    const float* __restrict__ inv_ny, const float* __restrict__ inv_nc,
    const float* __restrict__ cls_xy_w, const float* __restrict__ cls_xy_b,
    const float* __restrict__ cls_cy_w, const float* __restrict__ cls_cy_b,
    const float* __restrict__ xc_w, const float* __restrict__ xc_b,
    const int* __restrict__ use_cy_p, float* __restrict__ out) {
  const int wave = threadIdx.x >> 5, lane = threadIdx.x & 31;
  const int gw = blockIdx.x * 8 + wave;  // 0..159743
  const int b = gw / 624, j = gw % 624;
  const float *vh, *emb, *wv;
  float inv, bias;
  float* op;
  bool zero_out = false;
  if (j < 200) {
    vh = h_xy + (size_t)b * 1024; emb = y_emb + (size_t)j * 1024; inv = inv_ny[j];
    wv = cls_xy_w; bias = cls_xy_b[0]; op = out + (size_t)b * 200 + j;
  } else if (j < 400) {
    const int c = j - 200;
    vh = c_proj + (size_t)b * 1024; emb = y_emb + (size_t)c * 1024; inv = inv_ny[c];
    wv = cls_cy_w; bias = cls_cy_b[0]; op = out + 51200 + (size_t)b * 200 + c;
    zero_out = (*use_cy_p == 0);
  } else {
    const int r = j - 400;                // 0..223 = s*112 + n
    const int s = r / 112, n = r % 112;
    vh = h_xc + (size_t)b * 1024; emb = c_emb + (size_t)r * 1024; inv = inv_nc[r];
    wv = xc_w + (size_t)n * 1024; bias = xc_b[n];
    op = out + 102400 + (size_t)b * 224 + n * 2 + s;
  }
  float acc = 0.f;
  for (int h = lane * 4; h < 1024; h += 128) {
    float4 hv = *(const float4*)(vh + h);
    float4 ev = *(const float4*)(emb + h);
    float4 wf = *(const float4*)(wv + h);
    acc += fmaxf(hv.x * (1.f + ev.x * inv), 0.f) * wf.x;
    acc += fmaxf(hv.y * (1.f + ev.y * inv), 0.f) * wf.y;
    acc += fmaxf(hv.z * (1.f + ev.z * inv), 0.f) * wf.z;
    acc += fmaxf(hv.w * (1.f + ev.w * inv), 0.f) * wf.w;
  }
  for (int off = 16; off; off >>= 1) acc += __shfl_xor(acc, off, 32);
  if (lane == 0) *op = zero_out ? 0.0f : (acc + bias);
}

// ---------------------------------------------------------------------------
extern "C" void kernel_launch(void* const* d_in, const int* in_sizes, int n_in,
                              void* d_out, int out_size, void* d_ws, size_t ws_size,
                              hipStream_t stream) {
  (void)in_sizes; (void)n_in; (void)out_size; (void)ws_size;
  const float* x        = (const float*)d_in[0];
  const int*   c_gt     = (const int*)  d_in[1];
  const float* y_emb    = (const float*)d_in[2];
  const float* c_emb    = (const float*)d_in[3];
  const float* w_xy     = (const float*)d_in[4];
  const float* b_xy     = (const float*)d_in[5];
  const float* w_xc     = (const float*)d_in[6];
  const float* b_xc     = (const float*)d_in[7];
  const float* cls_xy_w = (const float*)d_in[8];
  const float* cls_xy_b = (const float*)d_in[9];
  const float* cls_cy_w = (const float*)d_in[10];
  const float* cls_cy_b = (const float*)d_in[11];
  const float* xc_w     = (const float*)d_in[12];
  const float* xc_b     = (const float*)d_in[13];
  const float* cpw      = (const float*)d_in[14];
  const float* cpb      = (const float*)d_in[15];
  const int*   use_cy_p = (const int*)  d_in[17];
  float* out = (float*)d_out;

  float* ws     = (float*)d_ws;
  float* inv_ny = ws;            // 200  (pad to 256)
  float* inv_nc = ws + 256;      // 224  (pad to 512)
  float* qsum   = ws + 512;      // 1024
  float* hxy    = ws + 1536;     // 262144
  float* hxc    = ws + 263680;   // 262144
  float* cproj  = ws + 525824;   // 262144
  float* Dm     = ws + 787968;   // 114688
  float* Qm     = ws + 902656;   // 114688  (end: 1017344 floats ~ 4.07 MB)

  // Host-side JAX key derivation: key(42) = (0,42); split -> k1, k2.
  uint32_t a0, b0, a1, b1;
  threefry2x32(0u, 42u, 0u, 2u, &a0, &b0);
  threefry2x32(0u, 42u, 1u, 3u, &a1, &b1);
  // k1 = (a0, a1) for h_xy dropout, k2 = (b0, b1) for h_xc dropout.

  norms_kernel<<<53, 256, 0, stream>>>(y_emb, c_emb, inv_ny, inv_nc);
  fc1_wmma_kernel<<<dim3(16, 64, 2), 32, 0, stream>>>(
      x, w_xy, b_xy, w_xc, b_xc, hxy, hxc, a0, a1, b0, b1);
  pq_kernel<<<dim3(112, 128), 256, 0, stream>>>(c_emb, cpw, Dm, Qm);
  qsum_kernel<<<4, 256, 0, stream>>>(Qm, cpb, qsum);
  cproj_kernel<<<256, 256, 0, stream>>>(c_gt, Dm, qsum, cproj);
  energy_kernel<<<19968, 256, 0, stream>>>(
      hxy, hxc, cproj, y_emb, c_emb, inv_ny, inv_nc,
      cls_xy_w, cls_xy_b, cls_cy_w, cls_cy_b, xc_w, xc_b, use_cy_p, out);
}
